// SimpleSSM_55405078118790
// MI455X (gfx1250) — matrix operand order repaired and checked by reading, not compiled, not run on hardware
//
#include <hip/hip_runtime.h>
#include <hip/hip_bf16.h>

// ---------------------------------------------------------------------------
// SimpleSSM on MI455X (gfx1250, wave32, WMMA)
//   h = GELU(x @ W_in + b_in)            -> WMMA f16 GEMM, f32 accum
//   4x { h = LN(h + EMA(h, sigmoid(a))) }-> scan kernel + layernorm kernel
//   out = h[:, -1, :] @ W_out + b_out    -> tiny scalar GEMM
// ---------------------------------------------------------------------------

#define B_   32
#define S_   2048
#define DIN_ 64
#define DH_  1024
#define DOUT_ 64
#define M_   (B_ * S_)          // 65536 rows
#define NLAYERS_ 4

typedef __attribute__((ext_vector_type(16))) _Float16 v16h;
typedef __attribute__((ext_vector_type(8)))  _Float16 v8h;
typedef __attribute__((ext_vector_type(8)))  float    v8f;

union H16 { v16h v; v8h q[2]; _Float16 h[16]; };

// --------------------------- conversion kernels ----------------------------

__global__ __launch_bounds__(256) void cvt_x_f16(const float* __restrict__ x,
                                                 _Float16* __restrict__ xh, int n) {
    int i = blockIdx.x * 256 + threadIdx.x;
    if (i < n) xh[i] = (_Float16)x[i];
}

// W_in is [K=64][N=1024] row-major; store transposed Wt[N=1024][K=64] in f16
// so each WMMA B-fragment lane load is one contiguous 32B v16h.
__global__ __launch_bounds__(256) void cvt_w_f16t(const float* __restrict__ W,
                                                  _Float16* __restrict__ Wt) {
    int i = blockIdx.x * 256 + threadIdx.x;   // 65536
    int n = i >> 6;
    int k = i & 63;
    Wt[i] = (_Float16)W[k * DH_ + n];         // Wt[n*64 + k]
}

// ----------------------- GEMM + bias + exact GELU --------------------------
// One wave per 16x16 output tile. K=64 -> 2x v_wmma_f32_16x16x32_f16.
__global__ __launch_bounds__(256) void gemm_gelu_wmma(
    const _Float16* __restrict__ A,    // [M_][64]  (x in f16)
    const _Float16* __restrict__ Bt,   // [1024][64] (W_in^T in f16)
    const float*    __restrict__ bias, // [1024]
    float*          __restrict__ H)    // [M_][1024]
{
    const int lane   = threadIdx.x & 31;
    const int waveId = threadIdx.x >> 5;
    const int tile   = blockIdx.x * 8 + waveId;   // 262144 tiles total
    const int ntile  = tile & 63;                 // 64 N-tiles
    const int mtile  = tile >> 6;                 // 4096 M-tiles

    const int hs  = lane >> 4;                    // half-wave select
    const int l15 = lane & 15;

    // A fragment: row = M, lanes 0-15 hold K 0..15, lanes 16-31 hold K 16..31 (per 32-K step)
    const _Float16* Arow = A + (size_t)(mtile * 16 + l15) * DIN_;
    const int kbA = hs * 16;

    // B fragment: col = N, lanes 0-15 hold K 0..15, lanes 16-31 hold K 16..31
    const _Float16* Bcol = Bt + (size_t)(ntile * 16 + l15) * DIN_;
    const int kbB = hs * 16;

    v8f c = {};
#pragma unroll
    for (int s = 0; s < 2; ++s) {
        const int koff = s * 32;
        H16 ua;
        ua.q[0] = *(const v8h*)(Arow + koff + kbA);        // K = koff+kbA .. +7
        ua.q[1] = *(const v8h*)(Arow + koff + kbA + 8);    // K = koff+kbA+8 .. +15
        v16h bfrag = *(const v16h*)(Bcol + koff + kbB);    // 16 contiguous K values

        c = __builtin_amdgcn_wmma_f32_16x16x32_f16(
                /*neg_a=*/false, ua.v, /*neg_b=*/false, bfrag,
                /*c_mod=*/(short)0, c, /*reuse_a=*/false, /*reuse_b=*/false);
    }

    const int ncol = ntile * 16 + l15;
    const float bv = bias[ncol];
#pragma unroll
    for (int r = 0; r < 8; ++r) {
        const int m = mtile * 16 + r + hs * 8;
        float v = c[r] + bv;
        // exact GELU: 0.5*v*(1+erf(v/sqrt(2)))
        float g = 0.5f * v * (1.0f + erff(v * 0.70710678118654752f));
        H[(size_t)m * DH_ + ncol] = g;
    }
}

// ------------------------- EMA scan + residual (in place) ------------------
// One thread per (b, d) channel: 32768 threads = 1024 wave32s.
__global__ __launch_bounds__(256) void ema_add_kernel(
    float* __restrict__ H, const float* __restrict__ alphas, int layer)
{
    const int id = blockIdx.x * 256 + threadIdx.x;  // 0..32767
    const int d  = id & (DH_ - 1);
    const int b  = id >> 10;
    const float a = 1.0f / (1.0f + expf(-alphas[layer * DH_ + d]));
    const float oma = 1.0f - a;

    float e = 0.0f;
    float* p = H + (size_t)b * S_ * DH_ + d;
#pragma unroll 4
    for (int t = 0; t < S_; ++t) {
        __builtin_prefetch(p + (size_t)(t + 8) * DH_, 0, 1); // global_prefetch_b8
        const float hv = p[(size_t)t * DH_];
        e = fmaf(a, hv, oma * e);        // h_t = a*x_t + (1-a)*h_{t-1}
        p[(size_t)t * DH_] = hv + e;     // residual add, in place
    }
}

// ------------------------------ LayerNorm ----------------------------------
// One 256-thread block per (b, t) row of 1024 channels.
__global__ __launch_bounds__(256) void layernorm_kernel(
    float* __restrict__ H, const float* __restrict__ gamma,
    const float* __restrict__ beta)
{
    __shared__ float s_sum[8];
    __shared__ float s_sq[8];

    float* p = H + (size_t)blockIdx.x * DH_;
    float v[4];
    float sum = 0.0f, sq = 0.0f;
#pragma unroll
    for (int i = 0; i < 4; ++i) {
        v[i] = p[threadIdx.x + i * 256];
        sum += v[i];
        sq  += v[i] * v[i];
    }
    // wave32 reduction
#pragma unroll
    for (int off = 16; off > 0; off >>= 1) {
        sum += __shfl_xor(sum, off, 32);
        sq  += __shfl_xor(sq,  off, 32);
    }
    const int wid = threadIdx.x >> 5;
    if ((threadIdx.x & 31) == 0) { s_sum[wid] = sum; s_sq[wid] = sq; }
    __syncthreads();

    float tsum = 0.0f, tsq = 0.0f;
#pragma unroll
    for (int i = 0; i < 8; ++i) { tsum += s_sum[i]; tsq += s_sq[i]; }
    const float mu   = tsum * (1.0f / DH_);
    const float var  = tsq * (1.0f / DH_) - mu * mu;
    const float rinv = rsqrtf(var + 1e-5f);

#pragma unroll
    for (int i = 0; i < 4; ++i) {
        const int idx = threadIdx.x + i * 256;
        p[idx] = gamma[idx] * (v[i] - mu) * rinv + beta[idx];
    }
}

// --------------------------- output projection -----------------------------
// out[b, o] = h[b, S-1, :] . W_out[:, o] + b_out[o]; 2048 outputs.
__global__ __launch_bounds__(256) void outproj_kernel(
    const float* __restrict__ H, const float* __restrict__ Wout,
    const float* __restrict__ bout, float* __restrict__ out)
{
    const int id = blockIdx.x * 256 + threadIdx.x;  // 0..2047
    const int o  = id & (DOUT_ - 1);
    const int b  = id >> 6;
    const float* hrow = H + ((size_t)b * S_ + (S_ - 1)) * DH_;
    float acc = bout[o];
#pragma unroll 8
    for (int k = 0; k < DH_; ++k)
        acc = fmaf(hrow[k], Wout[k * DOUT_ + o], acc);
    out[id] = acc;
}

// ------------------------------- launcher ----------------------------------

extern "C" void kernel_launch(void* const* d_in, const int* in_sizes, int n_in,
                              void* d_out, int out_size, void* d_ws, size_t ws_size,
                              hipStream_t stream) {
    const float* x      = (const float*)d_in[0];  // [32,2048,64]
    const float* W_in   = (const float*)d_in[1];  // [64,1024]
    const float* b_in   = (const float*)d_in[2];  // [1024]
    const float* alphas = (const float*)d_in[3];  // [4,1024]
    const float* gamma  = (const float*)d_in[4];  // [1024]
    const float* beta   = (const float*)d_in[5];  // [1024]
    const float* W_out  = (const float*)d_in[6];  // [1024,64]
    const float* b_out  = (const float*)d_in[7];  // [64]
    float* out = (float*)d_out;                   // [32,1,64]

    // workspace layout (all 256B aligned)
    char* ws = (char*)d_ws;
    _Float16* xh = (_Float16*)ws;                        // 8,388,608 B
    _Float16* wt = (_Float16*)(ws + 8388608);            //   131,072 B
    float*    H  = (float*)(ws + 8388608 + 131072);      // 268,435,456 B

    const int NX = M_ * DIN_;  // 4,194,304

    cvt_x_f16<<<NX / 256, 256, 0, stream>>>(x, xh, NX);
    cvt_w_f16t<<<(DIN_ * DH_) / 256, 256, 0, stream>>>(W_in, wt);

    // 4096 M-tiles x 64 N-tiles, 8 wave-tiles per 256-thread block
    gemm_gelu_wmma<<<(M_ / 16) * (DH_ / 16) / 8, 256, 0, stream>>>(xh, wt, b_in, H);

    for (int l = 0; l < NLAYERS_; ++l) {
        ema_add_kernel<<<(B_ * DH_) / 256, 256, 0, stream>>>(H, alphas, l);
        layernorm_kernel<<<B_ * S_, 256, 0, stream>>>(H, gamma, beta);
    }

    outproj_kernel<<<(B_ * DOUT_) / 256, 256, 0, stream>>>(H, W_out, b_out, out);
}
